// TimeAwareMultiHeadAttention_22823456211511
// MI455X (gfx1250) — matrix operand order repaired and checked by reading, not compiled, not run on hardware
//
#include <hip/hip_runtime.h>
#include <hip/hip_bf16.h>

typedef __attribute__((ext_vector_type(16))) _Float16 v16h;
typedef __attribute__((ext_vector_type(8)))  float    v8f;

#define DM     256
#define L_SEQ  256
#define NB     8

// Convert 4x float4 (16 floats) into a v16h in element order.
__device__ inline v16h cvt16(float4 x0, float4 x1, float4 x2, float4 x3) {
  v16h r;
  r[0]=(_Float16)x0.x; r[1]=(_Float16)x0.y; r[2]=(_Float16)x0.z; r[3]=(_Float16)x0.w;
  r[4]=(_Float16)x1.x; r[5]=(_Float16)x1.y; r[6]=(_Float16)x1.z; r[7]=(_Float16)x1.w;
  r[8]=(_Float16)x2.x; r[9]=(_Float16)x2.y; r[10]=(_Float16)x2.z; r[11]=(_Float16)x2.w;
  r[12]=(_Float16)x3.x; r[13]=(_Float16)x3.y; r[14]=(_Float16)x3.z; r[15]=(_Float16)x3.w;
  return r;
}

// Out[M,256] = X[M,256] @ W[256,256]^T + bias[256]
// One wave computes one 16x16 tile; K=256 in 8 WMMA(16x16x32_f16) steps.
// A layout (ISA 7.12.2): lane m=lane&15; kb=(lane&16)?8:0; e<8 -> K=kb+e,
//   e>=8 -> K=kb+16+(e-8).  B layout: lane n=lane&15; same K mapping as A.
__global__ __launch_bounds__(32)
void gemm_xwt_wmma(const float* __restrict__ X, const float* __restrict__ W,
                   const float* __restrict__ bias, float* __restrict__ Out) {
  const int lane = threadIdx.x;
  const int mt = blockIdx.x;
  const int nt = blockIdx.y;
  const int mrow = mt * 16 + (lane & 15);
  const int ncol = nt * 16 + (lane & 15);
  const int hiA = (lane & 16) ? 8  : 0;
  const int hiB = (lane & 16) ? 8  : 0;
  const float* __restrict__ arow = X + (size_t)mrow * DM;
  const float* __restrict__ brow = W + (size_t)ncol * DM;

  v8f c = {};
#pragma unroll
  for (int k0 = 0; k0 < DM; k0 += 32) {
    const float* ap = arow + k0 + hiA;
    const float* bp = brow + k0 + hiB;
    float4 a0 = *(const float4*)(ap +  0);
    float4 a1 = *(const float4*)(ap +  4);
    float4 a2 = *(const float4*)(ap + 16);
    float4 a3 = *(const float4*)(ap + 20);
    float4 b0 = *(const float4*)(bp +  0);
    float4 b1 = *(const float4*)(bp +  4);
    float4 b2 = *(const float4*)(bp + 16);
    float4 b3 = *(const float4*)(bp + 20);
    v16h a = cvt16(a0, a1, a2, a3);
    v16h b = cvt16(b0, b1, b2, b3);
    c = __builtin_amdgcn_wmma_f32_16x16x32_f16(false, a, false, b,
                                               (short)0, c, false, false);
  }

  const float bcol = bias[ncol];
  const int mbase = mt * 16 + ((lane & 16) ? 8 : 0);
#pragma unroll
  for (int r = 0; r < 8; ++r) {
    Out[(size_t)(mbase + r) * DM + ncol] = c[r] + bcol;
  }
}

// Fused time-aware attention for one (b, q).
// Block = 256 threads = 8 waves.  float4 granularity:
//   c4 = (t&63)*4   : 4 consecutive channels per thread (head h = c4/32)
//   ks = t>>6       : one of 4 simultaneous k-rows
// Every wave's loads are a contiguous 512B (b128 per lane) from one row.
// Dot-reduction spans the 8-lane segment owning a head (3 shfl_xor steps).
__global__ __launch_bounds__(256)
void attn_fused(const float* __restrict__ projQ, const float* __restrict__ projK,
                const float* __restrict__ projV,
                const float* __restrict__ tK, const float* __restrict__ tV,
                float* __restrict__ attnO) {
  const int q = blockIdx.x;
  const int b = blockIdx.y;
  const int t = threadIdx.x;
  const int c4 = (t & 63) << 2;   // channel base
  const int ks = t >> 6;          // k sub-slot 0..3
  const int h  = c4 >> 5;         // head 0..7

  __shared__ float  qv[DM];
  __shared__ float  sc[NB][L_SEQ];
  __shared__ float4 red[256];

  qv[t] = projQ[((size_t)(b * L_SEQ + q)) * DM + t];
  __syncthreads();
  const float4 q4 = *(const float4*)&qv[c4];

  const float scale = 0.17677669529663687f;  // 1/sqrt(32)
  const float* __restrict__ Kb  = projK + (size_t)b * L_SEQ * DM;
  const float* __restrict__ Vb  = projV + (size_t)b * L_SEQ * DM;
  const size_t tbase = ((size_t)(b * L_SEQ + q)) * L_SEQ * DM;
  const float* __restrict__ tKb = tK + tbase;
  const float* __restrict__ tVb = tV + tbase;

  // ---- scores: sc[h][k] = scale * sum_d q[d] * (K[k][d] + tK[k][d]) ----
  for (int k0 = 0; k0 < L_SEQ; k0 += 4) {
    const int k = k0 + ks;
    const size_t off = (size_t)k * DM + c4;
    float4 kk = *(const float4*)(Kb + off);
    float4 tk = *(const float4*)(tKb + off);
    __builtin_prefetch((const void*)(tKb + off + 16 * DM), 0, 1);
    float p = q4.x * (kk.x + tk.x) + q4.y * (kk.y + tk.y)
            + q4.z * (kk.z + tk.z) + q4.w * (kk.w + tk.w);
    p += __shfl_xor(p, 4, 32);
    p += __shfl_xor(p, 2, 32);
    p += __shfl_xor(p, 1, 32);
    if ((t & 7) == 0) sc[h][k] = p * scale;
  }
  __syncthreads();

  // ---- softmax over k; wave w reduces head w's row ----
  {
    const int lane = t & 31;
    const int hw   = t >> 5;
    float vloc[L_SEQ / 32];
    float mx = -3.402823466e38f;
#pragma unroll
    for (int i = 0; i < L_SEQ / 32; ++i) {
      vloc[i] = sc[hw][lane + 32 * i];
      mx = fmaxf(mx, vloc[i]);
    }
#pragma unroll
    for (int o = 16; o > 0; o >>= 1) mx = fmaxf(mx, __shfl_xor(mx, o, 32));
    float s = 0.f;
#pragma unroll
    for (int i = 0; i < L_SEQ / 32; ++i) {
      vloc[i] = __expf(vloc[i] - mx);
      s += vloc[i];
    }
#pragma unroll
    for (int o = 16; o > 0; o >>= 1) s += __shfl_xor(s, o, 32);
    const float inv = 1.0f / s;
#pragma unroll
    for (int i = 0; i < L_SEQ / 32; ++i) sc[hw][lane + 32 * i] = vloc[i] * inv;
  }
  __syncthreads();

  // ---- output: out[c] = sum_k attn[h][k] * (V[k][c] + tV[k][c]) ----
  float4 acc = make_float4(0.f, 0.f, 0.f, 0.f);
  for (int k0 = 0; k0 < L_SEQ; k0 += 4) {
    const int k = k0 + ks;
    const size_t off = (size_t)k * DM + c4;
    float4 vv = *(const float4*)(Vb + off);
    float4 tv = *(const float4*)(tVb + off);
    __builtin_prefetch((const void*)(tVb + off + 16 * DM), 0, 1);
    const float a = sc[h][k];
    acc.x = fmaf(a, vv.x + tv.x, acc.x);
    acc.y = fmaf(a, vv.y + tv.y, acc.y);
    acc.z = fmaf(a, vv.z + tv.z, acc.z);
    acc.w = fmaf(a, vv.w + tv.w, acc.w);
  }
  red[t] = acc;
  __syncthreads();

  // combine the 4 k-slot partials; coalesced b128 store of the (b,q) row
  if (t < 64) {
    float4 r0 = red[t];
    float4 r1 = red[t + 64];
    float4 r2 = red[t + 128];
    float4 r3 = red[t + 192];
    float4 o;
    o.x = (r0.x + r1.x) + (r2.x + r3.x);
    o.y = (r0.y + r1.y) + (r2.y + r3.y);
    o.z = (r0.z + r1.z) + (r2.z + r3.z);
    o.w = (r0.w + r1.w) + (r2.w + r3.w);
    *(float4*)(attnO + ((size_t)(b * L_SEQ + q)) * DM + t * 4) = o;
  }
}

extern "C" void kernel_launch(void* const* d_in, const int* in_sizes, int n_in,
                              void* d_out, int out_size, void* d_ws, size_t ws_size,
                              hipStream_t stream) {
  (void)in_sizes; (void)n_in; (void)out_size; (void)ws_size;
  const float* Q  = (const float*)d_in[0];
  const float* K  = (const float*)d_in[1];
  const float* V  = (const float*)d_in[2];
  const float* tK = (const float*)d_in[3];
  const float* tV = (const float*)d_in[4];
  const float* Wq = (const float*)d_in[5];
  const float* bq = (const float*)d_in[6];
  const float* Wk = (const float*)d_in[7];
  const float* bk = (const float*)d_in[8];
  const float* Wv = (const float*)d_in[9];
  const float* bv = (const float*)d_in[10];
  const float* Wo = (const float*)d_in[11];
  const float* bo = (const float*)d_in[12];
  float* out = (float*)d_out;
  float* ws  = (float*)d_ws;

  const int M = NB * L_SEQ;               // 2048 rows
  const size_t MAT = (size_t)M * DM;      // 524288 floats (2 MB)
  float* projQ = ws;
  float* projK = ws + MAT;
  float* projV = ws + 2 * MAT;
  float* attnO = ws + 3 * MAT;

  dim3 gg(M / 16, DM / 16);               // 128 x 16 tiles, 1 wave each
  gemm_xwt_wmma<<<gg, 32, 0, stream>>>(Q, Wq, bq, projQ);
  gemm_xwt_wmma<<<gg, 32, 0, stream>>>(K, Wk, bk, projK);
  gemm_xwt_wmma<<<gg, 32, 0, stream>>>(V, Wv, bv, projV);

  attn_fused<<<dim3(L_SEQ, NB), 256, 0, stream>>>(projQ, projK, projV, tK, tV, attnO);

  gemm_xwt_wmma<<<gg, 32, 0, stream>>>(attnO, Wo, bo, out);
}
